// AttentionLayer_38190849196536
// MI455X (gfx1250) — matrix operand order repaired and checked by reading, not compiled, not run on hardware
//
#include <hip/hip_runtime.h>
#include <hip/hip_bf16.h>

// CDNA5 (gfx1250) — wave32, WMMA 16x16x32 f16->f32.
typedef __attribute__((ext_vector_type(16))) _Float16 v16h;
typedef __attribute__((ext_vector_type(8)))  float    v8f;

#define HDIM 128
#define HQ   64
#define DDIM 131
#define M3STR 132   // padded row stride for M3 so float4 loads stay 16B aligned
#define KMAX 16

__device__ __forceinline__ float4 ld4(const float* p) {
  return *reinterpret_cast<const float4*>(p);
}

__device__ __forceinline__ float wave_sum(float v) {
#pragma unroll
  for (int off = 16; off > 0; off >>= 1) v += __shfl_xor(v, off, 32);
  return v;
}

// ---------------------------------------------------------------------------
// Kernel A: fold the small weight matrices (all tiny; ~2 MFLOP total).
//   Ppack : P = Wq^T @ Wk (128x128) computed DIRECTLY into f16, pre-swizzled
//           into WMMA B-fragment order:
//             element ((t*4 + c)*32 + lane)*16 + j  holds
//             P[k][col] with col = t*16 + (lane&15),
//                            k   = c*32 + (lane>=16 ? 16 : 0) + j
//           so kernel B loads one contiguous v16h (32B) per lane per K-chunk.
//   r  [128]   = bq @ Wk
//   v  [128]   = Wq^T @ bk
//   c  scalar  = bq . bk
//   M3 [3x132] = Wo @ Wc   (row stride 132, cols 0..130 valid)
//   b3 [3]     = Wo @ bc + bo
// ---------------------------------------------------------------------------
__global__ void setup_fold_kernel(const float* __restrict__ Wq,
                                  const float* __restrict__ bq,
                                  const float* __restrict__ Wk,
                                  const float* __restrict__ bk,
                                  const float* __restrict__ Wc,
                                  const float* __restrict__ bc,
                                  const float* __restrict__ Wo,
                                  const float* __restrict__ bo,
                                  _Float16* __restrict__ Ppack,
                                  float* __restrict__ r,
                                  float* __restrict__ v,
                                  float* __restrict__ cs,
                                  float* __restrict__ M3,
                                  float* __restrict__ b3) {
  const int t = blockIdx.x * blockDim.x + threadIdx.x;
  const int stride = gridDim.x * blockDim.x;

  // P, straight into packed f16 B-fragment layout.
  for (int e = t; e < HDIM * HDIM; e += stride) {
    const int j    = e & 15;
    const int lane = (e >> 4) & 31;
    const int c    = (e >> 9) & 3;
    const int tile = e >> 11;
    const int col  = tile * 16 + (lane & 15);
    const int kk   = c * 32 + ((lane >= 16) ? 16 : 0) + j;
    float s = 0.0f;
    for (int d = 0; d < HQ; ++d)
      s = fmaf(Wq[d * HDIM + kk], Wk[d * HDIM + col], s);
    Ppack[e] = (_Float16)s;
  }
  for (int e = t; e < HDIM; e += stride) {
    float sr = 0.0f, sv = 0.0f;
    for (int d = 0; d < HQ; ++d) {
      sr = fmaf(bq[d], Wk[d * HDIM + e], sr);
      sv = fmaf(Wq[d * HDIM + e], bk[d], sv);
    }
    r[e] = sr;
    v[e] = sv;
  }
  if (t == 0) {
    float s = 0.0f;
    for (int d = 0; d < HQ; ++d) s = fmaf(bq[d], bk[d], s);
    cs[0] = s;
  }
  for (int e = t; e < 3 * DDIM; e += stride) {
    const int j = e / DDIM, q = e % DDIM;
    float s = 0.0f;
    for (int d = 0; d < DDIM; ++d) s = fmaf(Wo[j * DDIM + d], Wc[d * DDIM + q], s);
    M3[j * M3STR + q] = s;
  }
  if (t < 3) {
    float s = 0.0f;
    for (int d = 0; d < DDIM; ++d) s = fmaf(Wo[t * DDIM + d], bc[d], s);
    b3[t] = s + bo[t];
  }
}

// ---------------------------------------------------------------------------
// Kernel B: qk = x @ P + r  via v_wmma_f32_16x16x32_f16.
// Block = 256 threads = 8 waves; wave w owns output column tile w (16 cols),
// block b owns node-row tile b (16 rows). K=128 -> 4 WMMA steps.
// Fragment layouts used here (contiguous k order in A and B):
//   A (16x32 f16): lane<16 -> row=lane, K {k0+0..15};
//                  lane>=16 -> row=lane-16, K {k0+16..31}
//   B (32x16 f16): pre-swizzled in Ppack; one contiguous v16h per lane/chunk
//   C/D (16x16 f32): vgpr i -> M=i (lane<16) / M=i+8 (lane>=16), N=lane&15
// ---------------------------------------------------------------------------
__global__ void qk_gemm_kernel(const float* __restrict__ x,
                               const _Float16* __restrict__ Ppack,
                               const float* __restrict__ r,
                               float* __restrict__ qk,
                               int n) {
  const int lane     = threadIdx.x & 31;
  const int colTile  = threadIdx.x >> 5;          // 0..7
  const int nodeBase = blockIdx.x * 16;
  const int hi       = (lane >= 16) ? 1 : 0;
  const int col      = colTile * 16 + (lane & 15);

  // B fragments: contiguous 32B per lane per K-chunk (pre-swizzled).
  v16h bfrag[4];
#pragma unroll
  for (int c = 0; c < 4; ++c) {
    const size_t off = ((size_t)((colTile * 4 + c) * 32 + lane)) << 4;
    bfrag[c] = *reinterpret_cast<const v16h*>(Ppack + off);
  }

  int gRow = nodeBase + (lane & 15);
  if (gRow >= n) gRow = n - 1;                    // clamp tail (stores guarded)
  const float* xr = x + (size_t)gRow * HDIM;
  const int aOff = hi ? 8 : 0;

  v8f acc = {};
#pragma unroll
  for (int c = 0; c < 4; ++c) {
    v16h afrag;
#pragma unroll
    for (int hb = 0; hb < 2; ++hb) {
      const int k0 = c * 32 + aOff * 2 + hb * 8;
      const float4 a0 = ld4(xr + k0);
      const float4 a1 = ld4(xr + k0 + 4);
      afrag[hb * 8 + 0] = (_Float16)a0.x;
      afrag[hb * 8 + 1] = (_Float16)a0.y;
      afrag[hb * 8 + 2] = (_Float16)a0.z;
      afrag[hb * 8 + 3] = (_Float16)a0.w;
      afrag[hb * 8 + 4] = (_Float16)a1.x;
      afrag[hb * 8 + 5] = (_Float16)a1.y;
      afrag[hb * 8 + 6] = (_Float16)a1.z;
      afrag[hb * 8 + 7] = (_Float16)a1.w;
    }
    acc = __builtin_amdgcn_wmma_f32_16x16x32_f16(false, afrag, false, bfrag[c],
                                                 (short)0, acc, false, false);
  }

  const float rv = r[col];
  const int mOff = hi ? 8 : 0;
#pragma unroll
  for (int i = 0; i < 8; ++i) {
    const int gm = nodeBase + i + mOff;
    if (gm < n) qk[(size_t)gm * HDIM + col] = acc[i] + rv;
  }
}

// ---------------------------------------------------------------------------
// Kernel C: one wave32 per node. Lane l owns feature dims [4l, 4l+4).
// scores_k = (qk[n].x_nbr + s0)/scale * mask; softmax over K;
// cbar = sum_k attn_k * [x_nbr, rel]; out = pts + M3 @ cbar + b3.
// Gathers hit L2 (x table = 25.6 MB << 192 MB L2).
// ---------------------------------------------------------------------------
template <int KT>
__global__ void attn_gather_kernel(const float* __restrict__ pts,
                                   const float* __restrict__ x,
                                   const int*   __restrict__ nbr,
                                   const float* __restrict__ qk,
                                   const float* __restrict__ vvec,
                                   const float* __restrict__ cscal,
                                   const float* __restrict__ M3,
                                   const float* __restrict__ b3,
                                   float* __restrict__ out,
                                   int n, int kruntime) {
  const int lane = threadIdx.x & 31;
  const int node = blockIdx.x * (blockDim.x >> 5) + (threadIdx.x >> 5);
  if (node >= n) return;
  const int kk = (KT > 0) ? KT : kruntime;

  const float4 q4 = ld4(qk + (size_t)node * HDIM + lane * 4);
  const float4 xo = ld4(x + (size_t)node * HDIM + lane * 4);
  const float4 vv = ld4(vvec + lane * 4);
  const float s0 =
      wave_sum(xo.x * vv.x + xo.y * vv.y + xo.z * vv.z + xo.w * vv.w) + cscal[0];

  const float pOwn = (lane < 3) ? pts[(size_t)node * 3 + lane] : 0.0f;
  const float invScale = 1.0f / (8.000001f);  // sqrt(64) + 1e-6

  int   idxs[KMAX];
  float sc[KMAX];
#pragma unroll
  for (int e = 0; e < ((KT > 0) ? KT : KMAX); ++e) {
    if (KT == 0 && e >= kk) break;
    const int idx = nbr[(size_t)node * kk + e];
    idxs[e] = idx;
    const float4 xn = ld4(x + (size_t)idx * HDIM + lane * 4);
    float d = q4.x * xn.x + q4.y * xn.y + q4.z * xn.z + q4.w * xn.w;
    d = wave_sum(d);
    const float s = (d + s0) * invScale;
    sc[e] = (idx != 0) ? s : 0.0f;  // reference quirk: mask MULTIPLIES scores
  }

  float m = -3.0e38f;
#pragma unroll
  for (int e = 0; e < ((KT > 0) ? KT : KMAX); ++e) {
    if (KT == 0 && e >= kk) break;
    m = fmaxf(m, sc[e]);
  }
  float sum = 0.0f;
#pragma unroll
  for (int e = 0; e < ((KT > 0) ? KT : KMAX); ++e) {
    if (KT == 0 && e >= kk) break;
    sc[e] = __expf(sc[e] - m);
    sum += sc[e];
  }
  const float inv = 1.0f / sum;

  float4 cb = make_float4(0.0f, 0.0f, 0.0f, 0.0f);
  float relbar = 0.0f;
#pragma unroll
  for (int e = 0; e < ((KT > 0) ? KT : KMAX); ++e) {
    if (KT == 0 && e >= kk) break;
    const float a = sc[e] * inv;
    const int idx = idxs[e];
    const float4 xn = ld4(x + (size_t)idx * HDIM + lane * 4);
    cb.x = fmaf(a, xn.x, cb.x);
    cb.y = fmaf(a, xn.y, cb.y);
    cb.z = fmaf(a, xn.z, cb.z);
    cb.w = fmaf(a, xn.w, cb.w);
    if (lane < 3) relbar = fmaf(a, pts[(size_t)idx * 3 + lane] - pOwn, relbar);
  }

  float disp0 = 0.0f, disp1 = 0.0f, disp2 = 0.0f;
#pragma unroll
  for (int j = 0; j < 3; ++j) {
    const float* m3r = M3 + j * M3STR;
    const float4 mm = ld4(m3r + lane * 4);
    float p = mm.x * cb.x + mm.y * cb.y + mm.z * cb.z + mm.w * cb.w;
    if (lane < 3) p = fmaf(m3r[HDIM + lane], relbar, p);
    p = wave_sum(p) + b3[j];
    if (j == 0) disp0 = p;
    if (j == 1) disp1 = p;
    if (j == 2) disp2 = p;
  }
  if (lane < 3) {
    const float dsel = (lane == 0) ? disp0 : (lane == 1) ? disp1 : disp2;
    out[(size_t)node * 3 + lane] = pOwn + dsel;
  }
}

extern "C" void kernel_launch(void* const* d_in, const int* in_sizes, int n_in,
                              void* d_out, int out_size, void* d_ws, size_t ws_size,
                              hipStream_t stream) {
  const float* pts = (const float*)d_in[0];
  const float* x   = (const float*)d_in[1];
  const float* Wq  = (const float*)d_in[2];
  const float* bq  = (const float*)d_in[3];
  const float* Wk  = (const float*)d_in[4];
  const float* bk  = (const float*)d_in[5];
  const float* Wc  = (const float*)d_in[6];
  const float* bc  = (const float*)d_in[7];
  const float* Wo  = (const float*)d_in[8];
  const float* bo  = (const float*)d_in[9];
  const int* edges = (const int*)d_in[10];

  const int n = in_sizes[0] / 3;
  const int k = (n > 0) ? in_sizes[10] / (2 * n) : 0;
  const int* dst = edges + (size_t)n * k;  // second row of edge_index
  float* out = (float*)d_out;

  // Workspace layout (256B-aligned slots): Ppack, r, v, c, M3, b3, qk
  char* ws = (char*)d_ws;
  _Float16* Ppack = (_Float16*)(ws + 0);   // 128*128*2 = 32768 B
  float* r  = (float*)(ws + 32768);        // 512 B
  float* v  = (float*)(ws + 33280);        // 512 B
  float* cs = (float*)(ws + 33792);        // 4 B
  float* M3 = (float*)(ws + 34048);        // 3*132*4 = 1584 B
  float* b3 = (float*)(ws + 35840);        // 12 B
  float* qk = (float*)(ws + 36864);        // n*128*4 B

  if (n <= 0 || k <= 0) return;

  setup_fold_kernel<<<64, 256, 0, stream>>>(Wq, bq, Wk, bk, Wc, bc, Wo, bo,
                                            Ppack, r, v, cs, M3, b3);

  const int nodeTiles = (n + 15) / 16;
  qk_gemm_kernel<<<nodeTiles, 256, 0, stream>>>(x, Ppack, r, qk, n);

  const int wavesPerBlock = 8;  // 256 threads
  const int blocksC = (n + wavesPerBlock - 1) / wavesPerBlock;
  if (k == 15) {
    attn_gather_kernel<15><<<blocksC, 256, 0, stream>>>(pts, x, dst, qk, v, cs,
                                                        M3, b3, out, n, k);
  } else {
    attn_gather_kernel<0><<<blocksC, 256, 0, stream>>>(pts, x, dst, qk, v, cs,
                                                       M3, b3, out, n, k);
  }
}
